// T5Attention_10505490006063
// MI455X (gfx1250) — compile-verified, matrix operand order checked by reading, not run on hardware
//
#include <hip/hip_runtime.h>

// Problem constants (fixed by the reference)
#define BSZ   4
#define SEQ   2048
#define DM    1024
#define NH    16
#define HD    64
#define MTOT  (BSZ * SEQ)      // 8192
#define NBUCK 32

typedef __attribute__((ext_vector_type(16))) __bf16 v16bf;
typedef __attribute__((ext_vector_type(8)))  __bf16 v8bf;
typedef __attribute__((ext_vector_type(8)))  float  v8f;
typedef int v4i __attribute__((vector_size(4 * sizeof(int))));

// ---------------------------------------------------------------------------
// gfx1250 async global->LDS copy (ASYNCcnt path), guarded so compile can't fail
// if the toolchain lacks the builtins; fallback is a register round-trip.
// Builtin signature (per hipcc diagnostic): (v4i addrspace(1)*, v4i addrspace(3)*,
// imm offset, imm cpol).
// ---------------------------------------------------------------------------
#if defined(__has_builtin)
#if __has_builtin(__builtin_amdgcn_global_load_async_to_lds_b128) && \
    __has_builtin(__builtin_amdgcn_s_wait_asynccnt)
#define USE_ASYNC_LDS 1
#endif
#endif
#ifndef USE_ASYNC_LDS
#define USE_ASYNC_LDS 0
#endif

static __device__ __forceinline__ void cp16_g2lds(void* lds_dst, const void* gsrc) {
#if USE_ASYNC_LDS
  __builtin_amdgcn_global_load_async_to_lds_b128(
      (__attribute__((address_space(1))) v4i*)(void*)gsrc,
      (__attribute__((address_space(3))) v4i*)lds_dst, 0, 0);
#else
  *(v8bf*)lds_dst = *(const v8bf*)gsrc;
#endif
}
static __device__ __forceinline__ void cp_wait() {
#if USE_ASYNC_LDS
  __builtin_amdgcn_s_wait_asynccnt(0);
#endif
}

// ---------------------------------------------------------------------------
// 16-bit A/B fragment loader for V_WMMA_F32_16X16X32_BF16.
// Per-lane: 8 contiguous bf16 at [row, kbase..kbase+7] and 8 at [row, kbase+16..+23].
// kbase = kstep + 8*(lane>>4). Works for global and LDS pointers.
// ---------------------------------------------------------------------------
static __device__ __forceinline__ v16bf load_frag(const __bf16* __restrict__ base,
                                                  int stride, int row, int kbase) {
  const __bf16* p = base + (size_t)row * stride + kbase;
  v8bf lo = *(const v8bf*)(p);
  v8bf hi = *(const v8bf*)(p + 16);
  return __builtin_shufflevector(lo, hi, 0, 1, 2, 3, 4, 5, 6, 7,
                                         8, 9, 10, 11, 12, 13, 14, 15);
}

static __device__ __forceinline__ v8f wmma_bf16(v16bf a, v16bf b, v8f c) {
  return __builtin_amdgcn_wmma_f32_16x16x32_bf16(false, a, false, b, (short)0, c,
                                                 false, false);
}

// ---------------------------------------------------------------------------
// fp32 -> bf16 conversions
// ---------------------------------------------------------------------------
__global__ void f32_to_bf16_kernel(const float* __restrict__ in,
                                   __bf16* __restrict__ out, int n) {
  int i = blockIdx.x * blockDim.x + threadIdx.x;
  if (i < n) out[i] = (__bf16)in[i];
}

// out[c][r] = in[r][c]  (weights stored transposed: B-frags read contiguously)
__global__ void f32_to_bf16_T_kernel(const float* __restrict__ in,
                                     __bf16* __restrict__ out, int rows, int cols) {
  int i = blockIdx.x * blockDim.x + threadIdx.x;
  if (i < rows * cols) {
    int r = i / cols, c = i % cols;
    out[(size_t)c * rows + r] = (__bf16)in[i];
  }
}

// ---------------------------------------------------------------------------
// T5 relative-position bias table: biasTab[h][delta + SEQ-1], delta = k - q
// ---------------------------------------------------------------------------
__global__ void bias_table_kernel(const float* __restrict__ rel_bias,
                                  float* __restrict__ biasTab) {
  int i = blockIdx.x * blockDim.x + threadIdx.x;
  const int nd = 2 * SEQ - 1;
  if (i >= nd * NH) return;
  int d = i / NH;
  int h = i % NH;
  int delta = d - (SEQ - 1);
  const int half_buckets = NBUCK / 2;     // 16
  const int max_exact = half_buckets / 2; // 8
  int bucket = (delta > 0) ? half_buckets : 0;
  int rel = delta < 0 ? -delta : delta;
  int v;
  if (rel < max_exact) {
    v = rel;
  } else {
    float rf = fmaxf((float)rel, 1.0f);
    int lg = (int)(__logf(rf / (float)max_exact) / __logf(128.0f / (float)max_exact)
                   * (float)(half_buckets - max_exact));
    v = min(max_exact + lg, half_buckets - 1);
  }
  bucket += v;
  biasTab[(size_t)h * nd + d] = rel_bias[bucket * NH + h];
}

// ---------------------------------------------------------------------------
// Tiled bf16 GEMM:  C[M,N] = A[M,K] * Bt[N,K]^T, f32 accumulate in WMMA.
// Block = 128 threads (4 waves). Block tile 128(M) x 64(N); each wave 32x64
// (2 M-tiles x 4 N-tiles = 8 WMMA per k-step on one staged B tile).
// B tile staged through LDS with async-to-LDS when available.
// OUT_MODE: 0 = bf16 row-major, 1 = f32 row-major, 2 = bf16 scattered to
//           Vt[b][h][d][s] (transposed per-head V for the P*V WMMA operand).
// ---------------------------------------------------------------------------
template <int OUT_MODE>
__global__ __launch_bounds__(128) void gemm_bf16_kernel(
    const __bf16* __restrict__ A, const __bf16* __restrict__ Bt,
    void* __restrict__ Cout, int Mm, int Nn, int Kk) {
  __shared__ __bf16 Bs[64][32];

  const int tid  = threadIdx.x;
  const int wave = tid >> 5;
  const int lane = tid & 31;
  const int lr   = lane & 15;
  const int half = lane >> 4;
  const int m0 = blockIdx.y * 128 + wave * 32;
  const int n0 = blockIdx.x * 64;

  v8f acc[2][4] = {};

  for (int kk = 0; kk < Kk; kk += 32) {
    // Stage Bt tile [64 rows(n)][32 k] into LDS: 16 bf16 (two b128) per thread.
    {
      int row = tid >> 1;
      int ko  = (tid & 1) * 16;
      const __bf16* src = Bt + (size_t)(n0 + row) * Kk + kk + ko;
      cp16_g2lds(&Bs[row][ko],     src);
      cp16_g2lds(&Bs[row][ko + 8], src + 8);
    }
    cp_wait();
    __syncthreads();

    __builtin_prefetch(A + (size_t)(m0 + lr) * Kk + kk + 64, 0, 0);

    v16bf af0 = load_frag(A, Kk, m0 + lr,      kk + 8 * half);
    v16bf af1 = load_frag(A, Kk, m0 + 16 + lr, kk + 8 * half);
#pragma unroll
    for (int t = 0; t < 4; ++t) {
      v16bf bfr = load_frag(&Bs[0][0], 32, t * 16 + lr, 8 * half);
      acc[0][t] = wmma_bf16(af0, bfr, acc[0][t]);
      acc[1][t] = wmma_bf16(af1, bfr, acc[1][t]);
    }
    __syncthreads();
  }

#pragma unroll
  for (int mt = 0; mt < 2; ++mt) {
#pragma unroll
    for (int t = 0; t < 4; ++t) {
#pragma unroll
      for (int r = 0; r < 8; ++r) {
        int m = m0 + mt * 16 + r + 8 * half;
        int n = n0 + t * 16 + lr;
        float v = acc[mt][t][r];
        if (OUT_MODE == 0) {
          ((__bf16*)Cout)[(size_t)m * Nn + n] = (__bf16)v;
        } else if (OUT_MODE == 1) {
          ((float*)Cout)[(size_t)m * Nn + n] = v;
        } else {
          int h = n >> 6, d = n & 63;
          int b = m >> 11, s = m & (SEQ - 1);
          ((__bf16*)Cout)[((size_t)(b * NH + h) * HD + d) * SEQ + s] = (__bf16)v;
        }
      }
    }
  }
}

// ---------------------------------------------------------------------------
// Flash attention with T5 bias. Grid = (SEQ/64, NH, BSZ), block = 128 (4 waves).
// K and V tiles for each 64-key block are staged once per block into LDS
// (async-to-LDS when available); all 4 waves (64 q-rows) consume them.
// Bias: T5 buckets saturate at |delta|>=91, so far-from-diagonal key blocks
// take a wave-uniform constant-bias fast path (no table gathers).
// ---------------------------------------------------------------------------
__global__ __launch_bounds__(128) void attn_kernel(
    const __bf16* __restrict__ Q, const __bf16* __restrict__ Kmat,
    const __bf16* __restrict__ Vt, const float* __restrict__ biasTab,
    const float* __restrict__ mask, __bf16* __restrict__ Ob) {
  __shared__ __bf16 Ks[64][64];      // [k_local][d]
  __shared__ __bf16 Vs[64][64];      // [d][k_local]
  __shared__ __bf16 Pst[4][16][64];  // wave-private P tiles (C->A relayout)

  const int tid  = threadIdx.x;
  const int wave = tid >> 5;
  const int lane = tid & 31;
  const int lr   = lane & 15;
  const int half = lane >> 4;
  const int b  = blockIdx.z;
  const int h  = blockIdx.y;
  const int q0 = blockIdx.x * 64 + wave * 16;

  const __bf16* Qbase = Q    + (size_t)(b * SEQ) * DM + h * HD;
  const __bf16* Kbase = Kmat + (size_t)(b * SEQ) * DM + h * HD;
  const __bf16* Vbase = Vt   + (size_t)(b * NH + h) * HD * SEQ;
  const float*  biasH = biasTab + (size_t)h * (2 * SEQ - 1);
  const float*  maskB = mask + b * SEQ;

  // Saturated-bucket bias constants (delta <= -91 -> bucket 15; >= +91 -> 31)
  const float farNeg = biasH[0];
  const float farPos = biasH[2 * (SEQ - 1)];

  // Q fragments are loop-invariant (HEAD_DIM = 64 = 2 k-steps)
  v16bf qa0 = load_frag(Qbase + (size_t)q0 * DM, DM, lr, 0  + 8 * half);
  v16bf qa1 = load_frag(Qbase + (size_t)q0 * DM, DM, lr, 32 + 8 * half);

  float mi[8], li[8];
#pragma unroll
  for (int r = 0; r < 8; ++r) { mi[r] = -1e30f; li[r] = 0.0f; }
  v8f oacc[4] = {};

  const int srow = tid >> 1;          // staging: row this thread copies
  const int soff = (tid & 1) * 32;    // staging: 32-element (64 B) half-row

  for (int kb = 0; kb < SEQ; kb += 64) {
    // ---- stage K (row=key, col=d) and V^T (row=d, col=key) tiles in LDS ----
    {
      const __bf16* ksrc = Kbase + (size_t)(kb + srow) * DM + soff;
      cp16_g2lds(&Ks[srow][soff],      ksrc);
      cp16_g2lds(&Ks[srow][soff + 8],  ksrc + 8);
      cp16_g2lds(&Ks[srow][soff + 16], ksrc + 16);
      cp16_g2lds(&Ks[srow][soff + 24], ksrc + 24);
      const __bf16* vsrc = Vbase + (size_t)srow * SEQ + kb + soff;
      cp16_g2lds(&Vs[srow][soff],      vsrc);
      cp16_g2lds(&Vs[srow][soff + 8],  vsrc + 8);
      cp16_g2lds(&Vs[srow][soff + 16], vsrc + 16);
      cp16_g2lds(&Vs[srow][soff + 24], vsrc + 24);
    }
    cp_wait();
    __syncthreads();

    // ---- scores S = Q * K^T (K frags from LDS, shared by all 4 waves) ----
    v8f sc[4] = {};
#pragma unroll
    for (int t = 0; t < 4; ++t) {
      v16bf kf0 = load_frag(&Ks[0][0], 64, t * 16 + lr, 0  + 8 * half);
      v16bf kf1 = load_frag(&Ks[0][0], 64, t * 16 + lr, 32 + 8 * half);
      sc[t] = wmma_bf16(qa0, kf0, sc[t]);
      sc[t] = wmma_bf16(qa1, kf1, sc[t]);
    }

    // ---- + rel-pos bias + additive mask (no 1/sqrt(d) in T5) ----
    // delta range for this (wave, key block): [kb-(q0+15), kb+63-q0]
    if (kb - (q0 + 15) >= 91) {          // fully saturated positive side
#pragma unroll
      for (int t = 0; t < 4; ++t) {
        float ma = (1.0f - maskB[kb + t * 16 + lr]) * -1e9f + farPos;
        sc[t] += ma;                     // splat add
      }
    } else if (kb + 63 - q0 <= -91) {    // fully saturated negative side
#pragma unroll
      for (int t = 0; t < 4; ++t) {
        float ma = (1.0f - maskB[kb + t * 16 + lr]) * -1e9f + farNeg;
        sc[t] += ma;
      }
    } else {                             // near-diagonal: table gather
#pragma unroll
      for (int t = 0; t < 4; ++t) {
        int k = kb + t * 16 + lr;
        float ma = (1.0f - maskB[k]) * -1e9f;
        int idx = k - (q0 + 8 * half) + (SEQ - 1);
#pragma unroll
        for (int r = 0; r < 8; ++r)
          sc[t][r] += biasH[idx - r] + ma;
      }
    }

    // ---- online softmax (row reductions across 16-lane halves) ----
#pragma unroll
    for (int r = 0; r < 8; ++r) {
      float mx = fmaxf(fmaxf(sc[0][r], sc[1][r]), fmaxf(sc[2][r], sc[3][r]));
#pragma unroll
      for (int off = 1; off < 16; off <<= 1)
        mx = fmaxf(mx, __shfl_xor(mx, off, 32));
      float mnew  = fmaxf(mi[r], mx);
      float alpha = __expf(mi[r] - mnew);
      float rs = 0.0f;
#pragma unroll
      for (int t = 0; t < 4; ++t) {
        float p = __expf(sc[t][r] - mnew);
        sc[t][r] = p;
        rs += p;
      }
#pragma unroll
      for (int off = 1; off < 16; off <<= 1)
        rs += __shfl_xor(rs, off, 32);
      li[r] = li[r] * alpha + rs;
      mi[r] = mnew;
#pragma unroll
      for (int t = 0; t < 4; ++t) oacc[t][r] *= alpha;
    }

    // ---- P: C-layout -> A-layout via wave-private LDS ----
#pragma unroll
    for (int t = 0; t < 4; ++t)
#pragma unroll
      for (int r = 0; r < 8; ++r)
        Pst[wave][r + 8 * half][t * 16 + lr] = (__bf16)sc[t][r];

    v16bf pf0 = load_frag(&Pst[wave][0][0], 64, lr, 0  + 8 * half);
    v16bf pf1 = load_frag(&Pst[wave][0][0], 64, lr, 32 + 8 * half);
    // ---- O += P * V  (V^T frags from LDS) ----
#pragma unroll
    for (int t = 0; t < 4; ++t) {
      v16bf vf0 = load_frag(&Vs[0][0], 64, t * 16 + lr, 0  + 8 * half);
      v16bf vf1 = load_frag(&Vs[0][0], 64, t * 16 + lr, 32 + 8 * half);
      oacc[t] = wmma_bf16(pf0, vf0, oacc[t]);
      oacc[t] = wmma_bf16(pf1, vf1, oacc[t]);
    }
    __syncthreads();   // protect Ks/Vs before next block's staging
  }

  // ---- normalize and store bf16 [b][s][h*64+d] ----
#pragma unroll
  for (int r = 0; r < 8; ++r) {
    int q = q0 + r + 8 * half;
    float inv = 1.0f / li[r];
#pragma unroll
    for (int t = 0; t < 4; ++t) {
      int d = t * 16 + lr;
      Ob[(size_t)(b * SEQ + q) * DM + h * HD + d] = (__bf16)(oacc[t][r] * inv);
    }
  }
}

// ---------------------------------------------------------------------------
// Launch
// ---------------------------------------------------------------------------
extern "C" void kernel_launch(void* const* d_in, const int* in_sizes, int n_in,
                              void* d_out, int out_size, void* d_ws, size_t ws_size,
                              hipStream_t stream) {
  (void)in_sizes; (void)n_in; (void)out_size; (void)ws_size;
  const float* hs   = (const float*)d_in[0];
  const float* mask = (const float*)d_in[1];
  const float* Wq   = (const float*)d_in[2];
  const float* Wk   = (const float*)d_in[3];
  const float* Wv   = (const float*)d_in[4];
  const float* Wo   = (const float*)d_in[5];
  const float* rb   = (const float*)d_in[6];

  char* ws = (char*)d_ws;
  size_t off = 0;
  auto alloc = [&](size_t bytes) -> char* {
    char* p = ws + off;
    off += (bytes + 255) & ~(size_t)255;
    return p;
  };

  __bf16* Xb   = (__bf16*)alloc((size_t)MTOT * DM * 2);   // hidden, bf16
  __bf16* Wqb  = (__bf16*)alloc((size_t)DM * DM * 2);     // W^T, bf16
  __bf16* Wkb  = (__bf16*)alloc((size_t)DM * DM * 2);
  __bf16* Wvb  = (__bf16*)alloc((size_t)DM * DM * 2);
  __bf16* Wob  = (__bf16*)alloc((size_t)DM * DM * 2);
  __bf16* Qb   = (__bf16*)alloc((size_t)MTOT * DM * 2);   // [b][s][h*64+d]
  __bf16* Kb   = (__bf16*)alloc((size_t)MTOT * DM * 2);   // [b][s][h*64+d]
  __bf16* Vtb  = (__bf16*)alloc((size_t)MTOT * DM * 2);   // [b][h][d][s]
  __bf16* Ob   = (__bf16*)alloc((size_t)MTOT * DM * 2);   // attn out bf16
  float*  bias = (float*) alloc((size_t)NH * (2 * SEQ - 1) * 4);

  // 1) conversions
  {
    int n = MTOT * DM;
    f32_to_bf16_kernel<<<(n + 255) / 256, 256, 0, stream>>>(hs, Xb, n);
    int nw = DM * DM;
    dim3 g((nw + 255) / 256);
    f32_to_bf16_T_kernel<<<g, 256, 0, stream>>>(Wq, Wqb, DM, DM);
    f32_to_bf16_T_kernel<<<g, 256, 0, stream>>>(Wk, Wkb, DM, DM);
    f32_to_bf16_T_kernel<<<g, 256, 0, stream>>>(Wv, Wvb, DM, DM);
    f32_to_bf16_T_kernel<<<g, 256, 0, stream>>>(Wo, Wob, DM, DM);
  }
  // 2) bias table
  {
    int n = (2 * SEQ - 1) * NH;
    bias_table_kernel<<<(n + 255) / 256, 256, 0, stream>>>(rb, bias);
  }
  // 3) QKV projections (WMMA GEMMs)
  {
    dim3 grid(DM / 64, MTOT / 128);  // (16, 64)
    gemm_bf16_kernel<0><<<grid, 128, 0, stream>>>(Xb, Wqb, Qb,  MTOT, DM, DM);
    gemm_bf16_kernel<0><<<grid, 128, 0, stream>>>(Xb, Wkb, Kb,  MTOT, DM, DM);
    gemm_bf16_kernel<2><<<grid, 128, 0, stream>>>(Xb, Wvb, Vtb, MTOT, DM, DM);
  }
  // 4) flash attention with T5 bias
  {
    dim3 grid(SEQ / 64, NH, BSZ);    // (32, 16, 4)
    attn_kernel<<<grid, 128, 0, stream>>>(Qb, Kb, Vtb, bias, mask, Ob);
  }
  // 5) output projection -> f32 d_out
  {
    dim3 grid(DM / 64, MTOT / 128);
    gemm_bf16_kernel<1><<<grid, 128, 0, stream>>>(Ob, Wob, d_out, MTOT, DM, DM);
  }
}
